// MultiHeadAttention_21844203667922
// MI455X (gfx1250) — matrix operand order repaired and checked by reading, not compiled, not run on hardware
//
#include <hip/hip_runtime.h>

// ---------------------------------------------------------------------------
// MHA forward for MI455X (gfx1250, wave32, WMMA bf16 -> f32, TDM async tiles)
// B=2, S=2048, D=1024, H=16, DK=64
// ---------------------------------------------------------------------------

typedef __attribute__((ext_vector_type(16))) __bf16 v16bf;
typedef __attribute__((ext_vector_type(8)))  __bf16 v8bf;
typedef __attribute__((ext_vector_type(8)))  float  v8f;
typedef __attribute__((ext_vector_type(4)))  unsigned int v4u;
typedef __attribute__((ext_vector_type(8)))  int v8i;
typedef __attribute__((ext_vector_type(4)))  int v4i;

static constexpr int BB = 2, S = 2048, D = 1024, H = 16, DK = 64;
static constexpr int MROWS = BB * S;  // 4096

__device__ __forceinline__ v16bf cat8(v8bf lo, v8bf hi) {
  return __builtin_shufflevector(lo, hi, 0, 1, 2, 3, 4, 5, 6, 7,
                                         8, 9, 10, 11, 12, 13, 14, 15);
}

__device__ __forceinline__ v8f zero8() {
  v8f z;
#pragma unroll
  for (int e = 0; e < 8; ++e) z[e] = 0.0f;
  return z;
}

__device__ __forceinline__ v8f wmma_bf16(v16bf a, v16bf b, v8f c) {
  // (neg_a, A, neg_b, B, c_mod, C, reuse_a, reuse_b)
  return __builtin_amdgcn_wmma_f32_16x16x32_bf16(false, a, false, b,
                                                 (short)0, c, false, false);
}

#if __has_builtin(__builtin_amdgcn_tensor_load_to_lds) && \
    __has_builtin(__builtin_amdgcn_s_wait_tensorcnt)
#define USE_TDM 1
#else
#define USE_TDM 0
#endif

#if USE_TDM
// 2D tile DMA: global (row-major, stride0 elems of 2B) -> LDS, with LDS row
// padding of 16B after every 128B (matches a [rows][72] __bf16 layout).
// D# layout per CDNA5 ISA.
// This toolchain's builtin is the 6-arg form:
//   (uint32x4 g0, int32x8 g1, int32x4 g2, int32x4 g3, int32x8, i32 cpol)
__device__ __forceinline__ void tdm_load_2d(unsigned lds_off,
                                            const __bf16* gptr,
                                            unsigned tensor_d0,
                                            unsigned tensor_d1,
                                            unsigned long long stride0,
                                            unsigned tile_d0,
                                            unsigned tile_d1) {
  const unsigned long long ga = (unsigned long long)gptr;
  v4u g0;
  g0[0] = 1u;                                   // count=1, user mode
  g0[1] = lds_off;                              // lds_addr (bytes)
  g0[2] = (unsigned)(ga & 0xFFFFFFFFull);       // global_addr[31:0]
  g0[3] = (unsigned)((ga >> 32) & 0x1FFFFFFull) // global_addr[56:32]
          | (2u << 30);                         // type = 2 ("image")
  v8i g1;
  // data_size=1 (2B), pad_enable=1, pad_interval=4 (32 DWORDs = 128B),
  // pad_amount=3 (4 DWORDs = 16B)
  g1[0] = (int)((1u << 16) | (1u << 20) | (4u << 22) | (3u << 25));
  g1[1] = (int)((tensor_d0 & 0xFFFFu) << 16);             // dim0[15:0]
  g1[2] = (int)((tensor_d0 >> 16) | ((tensor_d1 & 0xFFFFu) << 16));
  g1[3] = (int)((tensor_d1 >> 16) | (tile_d0 << 16));     // tile_dim0
  g1[4] = (int)(tile_d1 & 0xFFFFu);                       // tile_dim1 (dim2=0)
  g1[5] = (int)(unsigned)(stride0 & 0xFFFFFFFFull);       // dim0_stride lo
  g1[6] = (int)(unsigned)((stride0 >> 32) & 0xFFFFull);   // stride hi, d1s=0
  g1[7] = 0;
  v4i z4 = {0, 0, 0, 0};            // groups 2/3 unused for 2D tiles
  v8i z8 = {0, 0, 0, 0, 0, 0, 0, 0};
  __builtin_amdgcn_tensor_load_to_lds(g0, g1, z4, z4, z8, 0);
}
#endif

// ---------------------------------------------------------------------------
// Projection GEMM:  Y = X(fp32 [4096,1024]) @ W(fp32 [1024,1024]) + bias
// MODE 0: Y bf16 in head layout [B,H,S,DK]
// MODE 1: Y bf16 in transposed head layout [B,H,DK,S]   (for V)
// ---------------------------------------------------------------------------
template <int MODE>
__global__ __launch_bounds__(256) void proj_gemm(const float* __restrict__ X,
                                                 const float* __restrict__ W,
                                                 const float* __restrict__ bias,
                                                 __bf16* __restrict__ Y) {
  __shared__ __align__(16) __bf16 lB[128][40];  // [n][k] transposed, 80B rows
  const int tid = threadIdx.x;
  const int lane = tid & 31, wave = tid >> 5;
  const int half = lane >> 4, l16 = lane & 15;
  const int n0 = blockIdx.x * 128;
  const int m0 = blockIdx.y * 64;
  const int mw = m0 + (wave & 3) * 16;
  const int nwl = (wave >> 2) * 64;

  v8f acc[4];
#pragma unroll
  for (int t = 0; t < 4; ++t) acc[t] = zero8();

  const int arow = mw + l16;

  for (int k0 = 0; k0 < D; k0 += 32) {
    __syncthreads();
    {  // stage W tile (32K x 128N) into LDS transposed as bf16
      const int base = tid * 16;
      const int kk = base >> 7;
      const int nn = base & 127;
      const float4* wp = (const float4*)&W[(size_t)(k0 + kk) * D + n0 + nn];
#pragma unroll
      for (int q4 = 0; q4 < 4; ++q4) {
        float4 w4 = wp[q4];
        lB[nn + q4 * 4 + 0][kk] = (__bf16)w4.x;
        lB[nn + q4 * 4 + 1][kk] = (__bf16)w4.y;
        lB[nn + q4 * 4 + 2][kk] = (__bf16)w4.z;
        lB[nn + q4 * 4 + 3][kk] = (__bf16)w4.w;
      }
      if (k0 + 32 < D)
        __builtin_prefetch(&W[(size_t)(k0 + 32 + kk) * D + n0 + nn], 0, 1);
    }
    __syncthreads();

    v16bf a;
#pragma unroll
    for (int g = 0; g < 2; ++g) {
      const float4* ap =
          (const float4*)&X[(size_t)arow * D + k0 + g * 16 + half * 8];
      float4 x0 = ap[0], x1 = ap[1];
      a[g * 8 + 0] = (__bf16)x0.x;
      a[g * 8 + 1] = (__bf16)x0.y;
      a[g * 8 + 2] = (__bf16)x0.z;
      a[g * 8 + 3] = (__bf16)x0.w;
      a[g * 8 + 4] = (__bf16)x1.x;
      a[g * 8 + 5] = (__bf16)x1.y;
      a[g * 8 + 6] = (__bf16)x1.z;
      a[g * 8 + 7] = (__bf16)x1.w;
    }

#pragma unroll
    for (int t = 0; t < 4; ++t) {
      const __bf16* bp = &lB[nwl + t * 16 + l16][half * 8];
      v16bf bf = cat8(*(const v8bf*)bp, *(const v8bf*)(bp + 16));
      acc[t] = wmma_bf16(a, bf, acc[t]);
    }
  }

#pragma unroll
  for (int t = 0; t < 4; ++t) {
    const int col = n0 + nwl + t * 16 + l16;
    const float bv = bias[col];
    const int h = col / DK, dk = col % DK;
#pragma unroll
    for (int e = 0; e < 8; ++e) {
      const int row = mw + e + half * 8;
      const int b = row / S, s = row % S;
      const float v = acc[t][e] + bv;
      size_t idx;
      if (MODE == 0)
        idx = ((size_t)(b * H + h) * S + s) * DK + dk;
      else
        idx = ((size_t)(b * H + h) * DK + dk) * S + s;
      Y[idx] = (__bf16)v;
    }
  }
}

// ---------------------------------------------------------------------------
// Flash attention per (b,h): online softmax over kv tiles of 64.
// Block: 256 thr = 8 waves; wave owns 16 q rows (q tile 128); grid (S/128,H,B).
// K/V tiles double-buffered in LDS via the Tensor Data Mover (TENSORcnt).
// qh,kh: [B,H,S,DK] bf16; vt: [B,H,DK,S] bf16; oh: [B,H,S,DK] bf16.
// ---------------------------------------------------------------------------
__global__ __launch_bounds__(256) void attn_flash(const __bf16* __restrict__ qh,
                                                  const __bf16* __restrict__ kh,
                                                  const __bf16* __restrict__ vt,
                                                  __bf16* __restrict__ oh) {
  __shared__ __align__(16) __bf16 lK[2][64][72];  // [kv][dk], 144B rows
  __shared__ __align__(16) __bf16 lV[2][64][72];  // [dk][kv], 144B rows
  __shared__ __align__(16) __bf16 lP[8][16][72];  // per-wave P tile
  const int tid = threadIdx.x;
  const int lane = tid & 31, wave = tid >> 5;
  const int half = lane >> 4, l16 = lane & 15;
  const int q0 = blockIdx.x * 128;
  const int bh = blockIdx.z * H + blockIdx.y;

  const __bf16* qb = qh + (size_t)bh * S * DK;
  const __bf16* kb = kh + (size_t)bh * S * DK;
  const __bf16* vb = vt + (size_t)bh * DK * S;
  __bf16* ob = oh + (size_t)bh * S * DK;

  const int qrow = q0 + wave * 16 + l16;

  // Q fragments (A layout) live in registers for all kv tiles
  v16bf qf[2];
#pragma unroll
  for (int c = 0; c < 2; ++c) {
    const __bf16* p = qb + (size_t)qrow * DK + c * 32 + half * 8;
    qf[c] = cat8(*(const v8bf*)p, *(const v8bf*)(p + 16));
  }

  float m[8], l[8];
  v8f accO[4];
#pragma unroll
  for (int e = 0; e < 8; ++e) {
    m[e] = -3.0e38f;
    l[e] = 0.0f;
  }
#pragma unroll
  for (int t = 0; t < 4; ++t) accO[t] = zero8();

#if USE_TDM
  if (wave == 0) {  // kick off tile 0 DMA (K and V)
    tdm_load_2d((unsigned)(unsigned long long)&lK[0][0][0], kb, DK, S, DK, 64,
                64);
    tdm_load_2d((unsigned)(unsigned long long)&lV[0][0][0], vb, S, DK, S, 64,
                64);
  }
#endif

  const int ntiles = S / 64;
  for (int it = 0; it < ntiles; ++it) {
    const int buf = it & 1;
    const int kv0 = it * 64;
#if USE_TDM
    if (wave == 0) __builtin_amdgcn_s_wait_tensorcnt(0);
    __syncthreads();
    if (wave == 0 && it + 1 < ntiles) {  // overlap next tile DMA with compute
      const int kvn = kv0 + 64;
      tdm_load_2d((unsigned)(unsigned long long)&lK[buf ^ 1][0][0],
                  kb + (size_t)kvn * DK, DK, S, DK, 64, 64);
      tdm_load_2d((unsigned)(unsigned long long)&lV[buf ^ 1][0][0], vb + kvn, S,
                  DK, S, 64, 64);
    }
#else
    // Fallback: cooperative synchronous staging of K and V tiles
    __syncthreads();
    {
      const int idx = tid * 16;
      const int r = idx >> 6, c0 = idx & 63;
      const __bf16* gk = kb + (size_t)(kv0 + r) * DK + c0;
      const __bf16* gv = vb + (size_t)r * S + kv0 + c0;
      *(v8bf*)&lK[buf][r][c0] = *(const v8bf*)gk;
      *(v8bf*)&lK[buf][r][c0 + 8] = *(const v8bf*)(gk + 8);
      *(v8bf*)&lV[buf][r][c0] = *(const v8bf*)gv;
      *(v8bf*)&lV[buf][r][c0 + 8] = *(const v8bf*)(gv + 8);
    }
    __syncthreads();
#endif

    // scores: 16(q) x 64(kv) per wave = 4 N-tiles x 2 K-chunks of WMMA
    v8f sc[4];
#pragma unroll
    for (int t = 0; t < 4; ++t) sc[t] = zero8();
#pragma unroll
    for (int t = 0; t < 4; ++t) {
#pragma unroll
      for (int c = 0; c < 2; ++c) {
        const __bf16* p = &lK[buf][t * 16 + l16][c * 32 + half * 8];
        v16bf kf = cat8(*(const v8bf*)p, *(const v8bf*)(p + 16));
        sc[t] = wmma_bf16(qf[c], kf, sc[t]);
      }
    }

    // scale 1/sqrt(DK) = 0.125
#pragma unroll
    for (int t = 0; t < 4; ++t)
#pragma unroll
      for (int e = 0; e < 8; ++e) sc[t][e] *= 0.125f;

    // online softmax across 16-lane column groups
    float alpha[8];
#pragma unroll
    for (int e = 0; e < 8; ++e) {
      float v = fmaxf(fmaxf(sc[0][e], sc[1][e]), fmaxf(sc[2][e], sc[3][e]));
#pragma unroll
      for (int msk = 1; msk < 16; msk <<= 1)
        v = fmaxf(v, __shfl_xor(v, msk, 32));
      const float mnew = fmaxf(m[e], v);
      alpha[e] = __expf(m[e] - mnew);
      m[e] = mnew;
    }
#pragma unroll
    for (int e = 0; e < 8; ++e) {
      float s = 0.0f;
#pragma unroll
      for (int t = 0; t < 4; ++t) {
        const float p = __expf(sc[t][e] - m[e]);
        sc[t][e] = p;
        s += p;
      }
#pragma unroll
      for (int msk = 1; msk < 16; msk <<= 1) s += __shfl_xor(s, msk, 32);
      l[e] = l[e] * alpha[e] + s;
    }
#pragma unroll
    for (int t = 0; t < 4; ++t)
#pragma unroll
      for (int e = 0; e < 8; ++e) accO[t][e] *= alpha[e];

    // P: C-layout f32 -> A-layout bf16 via per-wave LDS tile
    __syncthreads();
#pragma unroll
    for (int t = 0; t < 4; ++t)
#pragma unroll
      for (int e = 0; e < 8; ++e)
        lP[wave][e + half * 8][t * 16 + l16] = (__bf16)sc[t][e];
    __syncthreads();

    // O += P @ V : 2 K-chunks (kv) x 4 N-tiles (dk)
#pragma unroll
    for (int c = 0; c < 2; ++c) {
      const __bf16* pp = &lP[wave][l16][c * 32 + half * 8];
      v16bf pf = cat8(*(const v8bf*)pp, *(const v8bf*)(pp + 16));
#pragma unroll
      for (int t = 0; t < 4; ++t) {
        const __bf16* vp = &lV[buf][t * 16 + l16][c * 32 + half * 8];
        v16bf vf = cat8(*(const v8bf*)vp, *(const v8bf*)(vp + 16));
        accO[t] = wmma_bf16(pf, vf, accO[t]);
      }
    }
  }

  // normalize and store O tile
#pragma unroll
  for (int e = 0; e < 8; ++e) l[e] = 1.0f / l[e];
#pragma unroll
  for (int t = 0; t < 4; ++t) {
    const int dk = t * 16 + l16;
#pragma unroll
    for (int e = 0; e < 8; ++e) {
      const int row = q0 + wave * 16 + e + half * 8;
      ob[(size_t)row * DK + dk] = (__bf16)(accO[t][e] * l[e]);
    }
  }
}

// ---------------------------------------------------------------------------
// Output GEMM: out(fp32 [4096,1024]) = oh @ Wo + bo
// ---------------------------------------------------------------------------
__global__ __launch_bounds__(256) void out_gemm(const __bf16* __restrict__ A,
                                                const float* __restrict__ W,
                                                const float* __restrict__ bias,
                                                float* __restrict__ Y) {
  __shared__ __align__(16) __bf16 lB[128][40];
  const int tid = threadIdx.x;
  const int lane = tid & 31, wave = tid >> 5;
  const int half = lane >> 4, l16 = lane & 15;
  const int n0 = blockIdx.x * 128;
  const int m0 = blockIdx.y * 64;
  const int mw = m0 + (wave & 3) * 16;
  const int nwl = (wave >> 2) * 64;

  v8f acc[4];
#pragma unroll
  for (int t = 0; t < 4; ++t) acc[t] = zero8();

  const int arow = mw + l16;
  const int b = arow >> 11;
  const int s = arow & (S - 1);

  for (int k0 = 0; k0 < D; k0 += 32) {
    __syncthreads();
    {
      const int base = tid * 16;
      const int kk = base >> 7;
      const int nn = base & 127;
      const float4* wp = (const float4*)&W[(size_t)(k0 + kk) * D + n0 + nn];
#pragma unroll
      for (int q4 = 0; q4 < 4; ++q4) {
        float4 w4 = wp[q4];
        lB[nn + q4 * 4 + 0][kk] = (__bf16)w4.x;
        lB[nn + q4 * 4 + 1][kk] = (__bf16)w4.y;
        lB[nn + q4 * 4 + 2][kk] = (__bf16)w4.z;
        lB[nn + q4 * 4 + 3][kk] = (__bf16)w4.w;
      }
      if (k0 + 32 < D)
        __builtin_prefetch(&W[(size_t)(k0 + 32 + kk) * D + n0 + nn], 0, 1);
    }
    __syncthreads();

    v16bf a;
#pragma unroll
    for (int g = 0; g < 2; ++g) {
      const int col = k0 + g * 16 + half * 8;  // within one head
      const int h = col >> 6, dk = col & 63;
      const __bf16* ap = A + ((size_t)(b * H + h) * S + s) * DK + dk;
      v8bf part = *(const v8bf*)ap;
#pragma unroll
      for (int j = 0; j < 8; ++j) a[g * 8 + j] = part[j];
    }

#pragma unroll
    for (int t = 0; t < 4; ++t) {
      const __bf16* bp = &lB[nwl + t * 16 + l16][half * 8];
      v16bf bf = cat8(*(const v8bf*)bp, *(const v8bf*)(bp + 16));
      acc[t] = wmma_bf16(a, bf, acc[t]);
    }
  }

#pragma unroll
  for (int t = 0; t < 4; ++t) {
    const int col = n0 + nwl + t * 16 + l16;
    const float bv = bias[col];
#pragma unroll
    for (int e = 0; e < 8; ++e) {
      const int row = mw + e + half * 8;
      Y[(size_t)row * D + col] = acc[t][e] + bv;
    }
  }
}

// ---------------------------------------------------------------------------
extern "C" void kernel_launch(void* const* d_in, const int* in_sizes, int n_in,
                              void* d_out, int out_size, void* d_ws,
                              size_t ws_size, hipStream_t stream) {
  (void)in_sizes; (void)n_in; (void)out_size; (void)ws_size;
  const float* q  = (const float*)d_in[0];
  const float* k  = (const float*)d_in[1];
  const float* v  = (const float*)d_in[2];
  // d_in[3]: mask (int) — mathematically unused in the reference
  const float* Wq = (const float*)d_in[4];
  const float* bq = (const float*)d_in[5];
  const float* Wk = (const float*)d_in[6];
  const float* bk = (const float*)d_in[7];
  const float* Wv = (const float*)d_in[8];
  const float* bv = (const float*)d_in[9];
  const float* Wo = (const float*)d_in[10];
  const float* bo = (const float*)d_in[11];
  float* out = (float*)d_out;

  const size_t elems = (size_t)BB * H * S * DK;  // 4 Mi elements, 8 MiB each
  __bf16* qh = (__bf16*)d_ws;
  __bf16* kh = qh + elems;
  __bf16* vt = kh + elems;
  __bf16* oh = vt + elems;

  dim3 gGemm(D / 128, MROWS / 64);  // (8, 64)
  proj_gemm<0><<<gGemm, 256, 0, stream>>>(q, Wq, bq, qh);
  proj_gemm<0><<<gGemm, 256, 0, stream>>>(k, Wk, bk, kh);
  proj_gemm<1><<<gGemm, 256, 0, stream>>>(v, Wv, bv, vt);
  attn_flash<<<dim3(S / 128, H, BB), 256, 0, stream>>>(qh, kh, vt, oh);
  out_gemm<<<gGemm, 256, 0, stream>>>(oh, Wo, bo, out);
}
